// INR_39109972197497
// MI455X (gfx1250) — matrix operand order repaired and checked by reading, not compiled, not run on hardware
//
#include <hip/hip_runtime.h>

// ---------------------------------------------------------------------------
// Fused INR kernel for MI455X (gfx1250), wave32 + v_wmma_f32_16x16x32_f16
// + global_load_async_to_lds_b128 (ASYNCcnt) weight staging.
//
// Problem constants: b=1, c=8, ah=aw=7, h=w=32, N=50176, dims 720->128^3->12.
// Exploits: rel_coord == 0 (drop W0 rows 648..715), wgt == 0.25,
// branch-dependent layer0 input is rank-1: preact_i = F + s_p * v_i.
//
// Prep kernel converts all weights to f16 once into d_ws (K-permuted,
// transposed, chunk-tiled). Main kernel async-DMAs them into LDS; every WMMA
// fragment load is two 16-byte LDS reads (layouts per ISA 7.12.2).
// ---------------------------------------------------------------------------

typedef __attribute__((ext_vector_type(16))) _Float16 v16h;
typedef __attribute__((ext_vector_type(8)))  float    v8f;
typedef __attribute__((ext_vector_type(4)))  unsigned uint4v;   // 16B aligned

union Frag { unsigned u[8]; uint4v q[2]; v16h v; };

__device__ __forceinline__ unsigned short h2us(float a) {
  union { _Float16 h; unsigned short u; } z; z.h = (_Float16)a; return z.u;
}
__device__ __forceinline__ unsigned pk2(float a, float b) {
  union { _Float16 h[2]; unsigned u; } z;
  z.h[0] = (_Float16)a; z.h[1] = (_Float16)b; return z.u;
}

// Async raw 16B copy global -> LDS, tracked by ASYNCcnt (ISA 15.18.3 op 98).
// ldsByteOff is an offset into the workgroup's LDS allocation (dynamic LDS
// starts at 0: no static __shared__ in this TU).
__device__ __forceinline__ void asyncCopy16(unsigned ldsByteOff, const void* gsrc) {
  asm volatile("global_load_async_to_lds_b128 %0, %1, off"
               :: "v"(ldsByteOff), "v"(gsrc) : "memory");
}
__device__ __forceinline__ void waitAsync0() {
  asm volatile("s_wait_asynccnt 0" ::: "memory");
}

// ---- workspace (d_ws) half-element offsets ----
#define WS_W0   0        // 21 chunks x [n 0..127][kp 0..31]  (86016 halves)
#define WS_W1   86016    // [n 0..127][k 0..127]              (16384)
#define WS_W2   102400   // [n 0..127][k 0..127]              (16384)
#define WS_W3   118784   // [n 0..15][k 0..127]               (2048)

// ---- LDS byte offsets (total 162,688 B < 320 KB / workgroup) ----
#define OFF_SV     0        // 14688 u16: [ang 0..8][svr 0..5][svc 0..33][c 0..7]
#define OFF_OFFP   29376    // 96 i32: p -> ((ang*6+ki)*34+kj)*8
#define OFF_W0B    29760    // 2 x 128x40 f16 double-buffered W0 K-chunks (permuted K)
#define OFF_W1T    50240    // 128x136 f16, transposed [n][k]
#define OFF_W2T    85056    // 128x136 f16
#define OFF_W3T    119872   // 16x136  f16 (cols 12..15 zero)
#define OFF_VV     124224   // 4x128 f32 branch vectors v_i
#define OFF_B0     126272
#define OFF_B1     126784
#define OFF_B2     127296
#define OFF_B3     127808   // 16 f32
#define OFF_H      127872   // 8 waves x 16x136 f16 activation tiles
#define SMEM_BYTES 162688

#define WSTRIDE 136         // half-elements per row: 272 B, multiple of 16
#define W0STRIDE 40         // chunk rows: 80 B, multiple of 16

// A-fragment (16-bit A 16x32, ISA 7.12.2): two contiguous 16B runs.
__device__ __forceinline__ v16h loadA(const unsigned short* H, int stride,
                                      int m, int kbase, int hi) {
  Frag f;
  const unsigned short* base = H + m * stride + kbase + (hi << 3);
  f.q[0] = *(const uint4v*)(base);        // K = kbase + hi*8 + 0..7
  f.q[1] = *(const uint4v*)(base + 16);   // K = kbase + 16 + hi*8 + 0..7
  return f.v;
}
// B-fragment (16-bit B 32x16): two 16B runs per lane, same K order as A.
__device__ __forceinline__ v16h loadB(const unsigned short* W, int stride,
                                      int n, int kbase, int hi) {
  Frag f;
  const unsigned short* base = W + n * stride + kbase + (hi << 3);
  f.q[0] = *(const uint4v*)(base);
  f.q[1] = *(const uint4v*)(base + 16);
  return f.v;
}

__device__ __forceinline__ void gemm128(const unsigned short* H, const unsigned short* Wt,
                                        int l15, int hi, v8f acc[8]) {
#pragma unroll
  for (int kc = 0; kc < 4; ++kc) {
    v16h a = loadA(H, WSTRIDE, l15, kc * 32, hi);
#pragma unroll
    for (int t = 0; t < 8; ++t) {
      v16h b = loadB(Wt, WSTRIDE, t * 16 + l15, kc * 32, hi);
      acc[t] = __builtin_amdgcn_wmma_f32_16x16x32_f16(false, a, false, b,
                                                      (short)0, acc[t], false, false);
    }
  }
}

__device__ __forceinline__ void epiReluStore(v8f acc[8], const float* bias,
                                             unsigned short* H, int l15, int hi) {
#pragma unroll
  for (int t = 0; t < 8; ++t) {
    float bb = bias[t * 16 + l15];
#pragma unroll
    for (int r = 0; r < 8; ++r) {
      float v = acc[t][r] + bb;
      v = v > 0.f ? v : 0.f;
      H[(hi * 8 + r) * WSTRIDE + t * 16 + l15] = h2us(v);
    }
  }
}

// ---------------------------------------------------------------------------
// Prep: convert/permute/tile all weights to f16 blobs in d_ws (runs once per
// call, 236 blocks x 256 threads, one packed u32 pair-write per thread).
// ---------------------------------------------------------------------------
extern "C" __global__ void __launch_bounds__(256)
inr_prep(const float* __restrict__ W0, const float* __restrict__ W1,
         const float* __restrict__ W2, const float* __restrict__ W3,
         unsigned short* __restrict__ ws)
{
  int gid = blockIdx.x * 256 + threadIdx.x;      // 60416 total
  if (gid < 43008) {                             // W0 chunks, permuted K = p*8+c
    int kci = gid / 2048, rem = gid % 2048;
    int kpi = rem >> 7, n = rem & 127;
    int gk = kci * 32 + kpi * 2;
    int p = gk >> 3, c = gk & 7;
    float f0 = 0.f, f1 = 0.f;
    if (p < 81) {
      f0 = W0[(c * 81 + p) * 128 + n];
      f1 = W0[((c + 1) * 81 + p) * 128 + n];
    }
    *(unsigned*)(ws + WS_W0 + kci * 4096 + n * 32 + kpi * 2) = pk2(f0, f1);
  } else if (gid < 59392) {                      // W1 / W2 transposed [n][k]
    int q = gid - 43008;
    const float* W = (q < 8192) ? W1 : W2;
    unsigned wsOfs = (q < 8192) ? WS_W1 : WS_W2;
    q &= 8191;
    int kpi = q >> 7, n = q & 127, k2 = kpi * 2;
    *(unsigned*)(ws + wsOfs + n * 128 + k2) = pk2(W[k2 * 128 + n], W[(k2 + 1) * 128 + n]);
  } else {                                       // W3 [16][k], cols >=12 zero
    int q = gid - 59392;
    int kpi = q >> 4, n = q & 15, k2 = kpi * 2;
    float f0 = (n < 12) ? W3[k2 * 12 + n]       : 0.f;
    float f1 = (n < 12) ? W3[(k2 + 1) * 12 + n] : 0.f;
    *(unsigned*)(ws + WS_W3 + n * 128 + k2) = pk2(f0, f1);
  }
}

extern "C" __global__ void __launch_bounds__(256)
inr_fused(const float* __restrict__ inp,
          const float* __restrict__ W0,           // rows 716..719 for v_i
          const float* __restrict__ b0, const float* __restrict__ b1,
          const float* __restrict__ b2, const float* __restrict__ b3,
          const unsigned short* __restrict__ ws,
          float* __restrict__ out)
{
  extern __shared__ char smem[];
  unsigned short* SV  = (unsigned short*)(smem + OFF_SV);
  int*            offP= (int*)(smem + OFF_OFFP);
  unsigned short* W0B = (unsigned short*)(smem + OFF_W0B);
  unsigned short* W1T = (unsigned short*)(smem + OFF_W1T);
  unsigned short* W2T = (unsigned short*)(smem + OFF_W2T);
  unsigned short* W3T = (unsigned short*)(smem + OFF_W3T);
  float*          VV  = (float*)(smem + OFF_VV);
  float*          B0L = (float*)(smem + OFF_B0);
  float*          B1L = (float*)(smem + OFF_B1);
  float*          B2L = (float*)(smem + OFF_B2);
  float*          B3L = (float*)(smem + OFF_B3);
  unsigned short* HT  = (unsigned short*)(smem + OFF_H);

  const int tid  = threadIdx.x;
  const int lane = tid & 31;
  const int wave = tid >> 5;
  const int hi   = (lane >> 4) & 1;
  const int l15  = lane & 15;

  const int blk    = blockIdx.x;          // 392 = 49 angular * 8 row-blocks
  const int aIdx   = blk >> 3;
  const int rowblk = blk & 7;
  const int a1 = aIdx / 7, a2 = aIdx % 7;
  const int i0 = rowblk << 2;             // 4 spatial rows per block
  const int irw   = wave >> 1;            // wave's row within block (0..3)
  const int jbase = (wave & 1) << 4;      // wave's column half (0 or 16)
  const int i_pt  = i0 + irw;

  // ---------------- Async-DMA all weight tiles into LDS ----------------
#pragma unroll
  for (int it = 0; it < 8; ++it) {
    int task = tid + it * 256;            // 2048 x 16B tasks per matrix
    int n = task >> 4, seg = task & 15;
    asyncCopy16(OFF_W1T + n * 272 + seg * 16, ws + WS_W1 + n * 128 + seg * 8);
    asyncCopy16(OFF_W2T + n * 272 + seg * 16, ws + WS_W2 + n * 128 + seg * 8);
  }
  {
    int n = tid >> 4, seg = tid & 15;     // 256 x 16B tasks
    asyncCopy16(OFF_W3T + n * 272 + seg * 16, ws + WS_W3 + n * 128 + seg * 8);
  }
#pragma unroll
  for (int it = 0; it < 2; ++it) {        // W0 chunk 0: 512 x 16B tasks
    int task = tid + it * 256;
    int n = task >> 2, seg = task & 3;
    asyncCopy16(OFF_W0B + n * 80 + seg * 16, ws + WS_W0 + n * 32 + seg * 8);
  }

  // ---------------- Stage input sub-volume + small tables ----------------
  // SV[((ang*6+svr)*34+svc)*8 + c], channels innermost (16B runs per (ang,r,c))
  for (int e = tid; e < 14688; e += 256) {
    int svc = e % 34; int t  = e / 34;
    int cc  = t % 8;  int t2 = t / 8;
    int svr = t2 % 6; int ang = t2 / 6;
    int ka = ang / 3, kb = ang % 3;
    int gi = i0 - 1 + svr, gj = svc - 1;
    int aa1 = a1 + ka - 1, aa2 = a2 + kb - 1;
    float v = 0.f;
    if ((unsigned)aa1 < 7u && (unsigned)aa2 < 7u &&
        (unsigned)gi < 32u && (unsigned)gj < 32u)
      v = inp[(((cc * 7 + aa1) * 7 + aa2) * 32 + gi) * 32 + gj];
    SV[((ang * 6 + svr) * 34 + svc) * 8 + cc] = h2us(v);
  }
  // p -> sub-volume base offsets (p >= 81: any valid address; B rows are zero)
  if (tid < 96) {
    int off = 0;
    if (tid < 81) {
      int s = tid / 9, ang = tid % 9;
      int ki = s / 3, kj = s % 3;
      off = ((ang * 6 + ki) * 34 + kj) * 8;
    }
    offP[tid] = off;
  }
  if (tid < 128) {
    float wA = W0[716 * 128 + tid] + W0[717 * 128 + tid];
    float wB = W0[718 * 128 + tid] + W0[719 * 128 + tid];
    float P = 32.f;
#pragma unroll
    for (int br = 0; br < 4; ++br) { VV[br * 128 + tid] = P * wA + wB; P *= 32.f; }
    B0L[tid] = b0[tid];
    B1L[tid] = b1[tid];
    B2L[tid] = b2[tid];
  }
  if (tid < 16) B3L[tid] = (tid < 12) ? b3[tid] : 0.f;

  waitAsync0();
  __syncthreads();

  // ---------------- Layer 0: F = q_feat @ W0[0:648] + b0 (K permuted) ----------------
  const int basePt8 = (irw * 34 + jbase + l15) * 8;   // per-lane sub-volume base
  v8f F[8];
#pragma unroll
  for (int t = 0; t < 8; ++t) F[t] = (v8f){};

  for (int kci = 0; kci < 21; ++kci) {
    const unsigned short* buf = W0B + (kci & 1) * (128 * W0STRIDE);
    if (kci + 1 < 21) {                         // async-stage next chunk
      unsigned dstBase = OFF_W0B + (unsigned)(((kci + 1) & 1) * (128 * W0STRIDE * 2));
      const unsigned short* src = ws + WS_W0 + (kci + 1) * 4096;
#pragma unroll
      for (int it = 0; it < 2; ++it) {
        int task = tid + it * 256;
        int n = task >> 2, seg = task & 3;
        asyncCopy16(dstBase + n * 80 + seg * 16, src + n * 32 + seg * 8);
      }
      if (kci + 2 < 21)                         // gfx1250 global_prefetch_b8
        __builtin_prefetch(ws + WS_W0 + (kci + 2) * 4096 + tid * 16, 0, 0);
    }
    // A fragment: two aligned 16B runs (channels innermost), per documented layout
    int p0 = (kci << 2) + hi;                   // VGPRs 0..3: K = p0*8 + 0..7
    int p1 = p0 + 2;                            // VGPRs 4..7: K = p1*8 + 0..7
    Frag fa;
    fa.q[0] = *(const uint4v*)(SV + offP[p0] + basePt8);
    fa.q[1] = *(const uint4v*)(SV + offP[p1] + basePt8);
#pragma unroll
    for (int t = 0; t < 8; ++t) {
      v16h fb = loadB(buf, W0STRIDE, t * 16 + l15, 0, hi);
      F[t] = __builtin_amdgcn_wmma_f32_16x16x32_f16(false, fa.v, false, fb,
                                                    (short)0, F[t], false, false);
    }
    waitAsync0();
    __syncthreads();
  }
#pragma unroll
  for (int t = 0; t < 8; ++t) {
    float bb = B0L[t * 16 + l15];
#pragma unroll
    for (int r = 0; r < 8; ++r) F[t][r] += bb;
  }

  // ---------------- Branch loop: layers 1..3, 4 cell variants ----------------
  unsigned short* H = HT + wave * (16 * WSTRIDE);   // wave-private activation tile
  float s8[8];
#pragma unroll
  for (int r = 0; r < 8; ++r) {
    int m = hi * 8 + r;
    int j = jbase + m;
    float s = 1.f;
    if (i_pt == 0 && j < 4) s = (j < 2) ? 0.0625f : (2.f / 7.f);
    s8[r] = s;
  }

  v8f outAcc = (v8f){};
  for (int br = 0; br < 4; ++br) {
    // h0 = relu(F + s_p * v_br)  -> LDS tile (f16)
#pragma unroll
    for (int t = 0; t < 8; ++t) {
      float vb = VV[br * 128 + t * 16 + l15];
#pragma unroll
      for (int r = 0; r < 8; ++r) {
        float p = F[t][r] + s8[r] * vb;
        p = p > 0.f ? p : 0.f;
        H[(hi * 8 + r) * WSTRIDE + t * 16 + l15] = h2us(p);
      }
    }
    v8f acc[8];
#pragma unroll
    for (int t = 0; t < 8; ++t) acc[t] = (v8f){};
    gemm128(H, W1T, l15, hi, acc);              // h1 = relu(h0 @ W1 + b1)
    epiReluStore(acc, B1L, H, l15, hi);
#pragma unroll
    for (int t = 0; t < 8; ++t) acc[t] = (v8f){};
    gemm128(H, W2T, l15, hi, acc);              // h2 = relu(h1 @ W2 + b2)
    epiReluStore(acc, B2L, H, l15, hi);
#pragma unroll
    for (int kc = 0; kc < 4; ++kc) {            // outAcc += h2 @ W3 (all branches)
      v16h a = loadA(H, WSTRIDE, l15, kc * 32, hi);
      v16h b = loadB(W3T, WSTRIDE, l15, kc * 32, hi);
      outAcc = __builtin_amdgcn_wmma_f32_16x16x32_f16(false, a, false, b,
                                                      (short)0, outAcc, false, false);
    }
  }

  // ---------------- Epilogue: 0.25*sum + b3, PixelShuffle scatter ----------------
  if (l15 < 12) {
    int ch = l15 >> 2, rr = (l15 >> 1) & 1, ss = l15 & 1;
    float bb = B3L[l15];
    int base = ((ch * 7 + a1) * 7 + a2) * 4096 + (2 * i_pt + rr) * 64 + ss;
#pragma unroll
    for (int r = 0; r < 8; ++r) {
      int m = hi * 8 + r;
      int j = jbase + m;
      out[base + 2 * j] = 0.25f * outAcc[r] + bb;
    }
  }
}

extern "C" void kernel_launch(void* const* d_in, const int* in_sizes, int n_in,
                              void* d_out, int out_size, void* d_ws, size_t ws_size,
                              hipStream_t stream) {
  const float* inp = (const float*)d_in[0];
  const float* W0  = (const float*)d_in[1];
  const float* b0  = (const float*)d_in[2];
  const float* W1  = (const float*)d_in[3];
  const float* b1  = (const float*)d_in[4];
  const float* W2  = (const float*)d_in[5];
  const float* b2  = (const float*)d_in[6];
  const float* W3  = (const float*)d_in[7];
  const float* b3  = (const float*)d_in[8];
  float* outp = (float*)d_out;
  unsigned short* ws = (unsigned short*)d_ws;
  (void)in_sizes; (void)n_in; (void)out_size; (void)ws_size;

  inr_prep<<<dim3(236), dim3(256), 0, stream>>>(W0, W1, W2, W3, ws);

  (void)hipFuncSetAttribute((const void*)inr_fused,
                            hipFuncAttributeMaxDynamicSharedMemorySize, SMEM_BYTES);
  inr_fused<<<dim3(392), dim3(256), SMEM_BYTES, stream>>>(
      inp, W0, b0, b1, b2, b3, ws, outp);
}
